// SimCSELoss_49881750176193
// MI455X (gfx1250) — compile-verified, matrix operand order checked by reading, not run on hardware
//
#include <hip/hip_runtime.h>
#include <math.h>

typedef __attribute__((ext_vector_type(16))) _Float16 v16h;
typedef __attribute__((ext_vector_type(8)))  _Float16 v8h;
typedef __attribute__((ext_vector_type(8)))  float    v8f;

#define D 768
#define KT 24             // 768 / 32 k-tiles
#define WPB 4             // waves per block in GEMM kernel (share one B tile)
#define ROWP 776          // padded LDS row stride in halves (1552 B; 388 dw = 4 mod 64 banks)
#define TILE_H (16 * ROWP)          // halves per 16-row padded tile
#define TILE_B (TILE_H * 2)         // bytes per padded tile (24832)
#define LDS_BYTES (WPB * TILE_B)    // 99328 B: A-stage region, later reused for B dbl-buf
// 20 / T fold: logits = cos * 20 ; exp(logits - 20) = exp2(cos*C - C), C = 20*log2(e)
#define C20L2E 28.853900817779268f

// ---------------------------------------------------------------------------
// Kernel 1: per-row normalize to f16 + diagonal logit (dot(n1_i,n2_i)*20)
// ---------------------------------------------------------------------------
__global__ void k_normalize(const float* __restrict__ out,
                            _Float16* __restrict__ n1h,
                            _Float16* __restrict__ n2h,
                            float* __restrict__ diag) {
    const int i = blockIdx.x;
    const float* z1 = out + (size_t)i * (2 * D);
    const float* z2 = z1 + D;

    float x1[3], x2[3];
    float s11 = 0.f, s22 = 0.f, s12 = 0.f;
#pragma unroll
    for (int e = 0; e < 3; ++e) {
        const int idx = threadIdx.x + 256 * e;
        x1[e] = z1[idx];
        x2[e] = z2[idx];
        s11 = fmaf(x1[e], x1[e], s11);
        s22 = fmaf(x2[e], x2[e], s22);
        s12 = fmaf(x1[e], x2[e], s12);
    }
#pragma unroll
    for (int m = 1; m < 32; m <<= 1) {
        s11 += __shfl_xor(s11, m);
        s22 += __shfl_xor(s22, m);
        s12 += __shfl_xor(s12, m);
    }
    __shared__ float sm[3][8];
    const int wv = threadIdx.x >> 5;
    const int ln = threadIdx.x & 31;
    if (ln == 0) { sm[0][wv] = s11; sm[1][wv] = s22; sm[2][wv] = s12; }
    __syncthreads();
    if (threadIdx.x == 0) {
        float t0 = 0.f, t1 = 0.f, t2 = 0.f;
#pragma unroll
        for (int w = 0; w < 8; ++w) { t0 += sm[0][w]; t1 += sm[1][w]; t2 += sm[2][w]; }
        sm[0][0] = t0; sm[1][0] = t1; sm[2][0] = t2;
    }
    __syncthreads();
    const float nrm1 = fmaxf(sqrtf(sm[0][0]), 1e-8f);
    const float nrm2 = fmaxf(sqrtf(sm[1][0]), 1e-8f);
    const float inv1 = 1.0f / nrm1;
    const float inv2 = 1.0f / nrm2;
    if (threadIdx.x == 0)
        diag[i] = 20.0f * sm[2][0] * inv1 * inv2;   // logits-scale diagonal
#pragma unroll
    for (int e = 0; e < 3; ++e) {
        const int idx = threadIdx.x + 256 * e;
        n1h[(size_t)i * D + idx] = (_Float16)(x1[e] * inv1);
        n2h[(size_t)i * D + idx] = (_Float16)(x2[e] * inv2);
    }
}

// ---------------------------------------------------------------------------
// Kernel 2: fused f16-WMMA GEMM (n1 @ n2^T) + fixed-max softmax denominator.
//  - one wave per 16-row M-tile, 4 waves/block share each 16-col B tile
//  - A fragments register-resident (staged via LDS, then loaded to v16h[24])
//  - B tiles double-buffered in LDS via async global->LDS copies (ASYNCcnt)
//  - IGLP sched_group_barrier keeps DS reads one k-step ahead of WMMA
// WMMA 16-bit A/B per-lane slice: lane L: r=L&15, h=L>>4;
//   elems 0..7 = row[k0 + h*8 .. +8), elems 8..15 = row[k0 + 16 + h*8 .. +8)
// ---------------------------------------------------------------------------
__global__ void __launch_bounds__(WPB * 32)
k_gemm_lse(const _Float16* __restrict__ n1h,
           const _Float16* __restrict__ n2h,
           float* __restrict__ ssum,
           int njt) {
    extern __shared__ char dynlds[];

    const int wave  = threadIdx.x >> 5;
    const int lane  = threadIdx.x & 31;
    const int r     = lane & 15;
    const int h     = lane >> 4;
    const int mbase = (blockIdx.x * WPB + wave) * 16;

    union pack { v16h v; v8h p[2]; };

    // --- stage this wave's 16x768 A tile into padded LDS, then to registers
    _Float16* aw = (_Float16*)dynlds + wave * TILE_H;
    {
        const _Float16* gbase = n1h + (size_t)mbase * D;
#pragma unroll
        for (int row = 0; row < 16; ++row) {
#pragma unroll
            for (int it = 0; it < 3; ++it) {
                const int col = lane + 32 * it;              // 0..95 chunks of 8 halves
                *(v8h*)(aw + row * ROWP + col * 8) =
                    *(const v8h*)(gbase + (size_t)row * D + col * 8);
            }
        }
    }
    __syncthreads();

    v16h a[KT];
    {
        const int abase = r * ROWP + h * 8;
#pragma unroll
        for (int k = 0; k < KT; ++k) {
            pack u;
            u.p[0] = *(const v8h*)(aw + abase + k * 32);
            u.p[1] = *(const v8h*)(aw + abase + k * 32 + 16);
            a[k] = u.v;
        }
    }
    __syncthreads();   // A region free; reuse first 2*TILE_B bytes as B buffers

    const unsigned ldsbase = (unsigned)(uintptr_t)(void*)dynlds;
    const int srow = threadIdx.x >> 3;   // 0..15 : B row staged by this thread
    const int ssub = threadIdx.x & 7;    // 8 threads per row, 12 chunks each

    // async-stage B tile jt into buffer buf (16 x 768 halves, padded rows)
    auto stage = [&](int jt, int buf) {
        const _Float16* g = n2h + ((size_t)jt * 16 + srow) * D;
        const unsigned lrow = ldsbase + (unsigned)buf * TILE_B + (unsigned)srow * (ROWP * 2);
#pragma unroll
        for (int q = 0; q < 12; ++q) {
            const int col = (ssub + 8 * q) * 8;          // halves
            const unsigned laddr = lrow + (unsigned)col * 2;
            const _Float16* ga = g + col;
            asm volatile("global_load_async_to_lds_b128 %0, %1, off"
                         :
                         : "v"(laddr), "v"(ga)
                         : "memory");
        }
    };

    float s[8];
#pragma unroll
    for (int g = 0; g < 8; ++g) s[g] = 0.f;

    stage(0, 0);
    asm volatile("s_wait_asynccnt 0x0" ::: "memory");
    __syncthreads();

    const int bbase = r * ROWP + h * 8;   // per-lane B slice base (halves)

    for (int jt = 0; jt < njt; ++jt) {
        if (jt + 1 < njt) stage(jt + 1, (jt + 1) & 1);

        const _Float16* bw = (const _Float16*)(dynlds + (jt & 1) * TILE_B);
        v8f acc = {};
#pragma unroll
        for (int k = 0; k < KT; ++k) {
            pack b;
            b.p[0] = *(const v8h*)(bw + bbase + k * 32);
            b.p[1] = *(const v8h*)(bw + bbase + k * 32 + 16);
            acc = __builtin_amdgcn_wmma_f32_16x16x32_f16(
                false, a[k], false, b.v, (short)0, acc, false, false);
        }

        // IGLP pipeline: DS x4, (WMMA x1, DS x2) x23, WMMA x1
        // -> forces ds_load(k+1) above wmma(k); RA keeps two B fragments live
        __builtin_amdgcn_sched_group_barrier(0x100, 4, 0);   // 2 fragments of DS reads
#pragma unroll
        for (int k = 0; k < KT - 1; ++k) {
            __builtin_amdgcn_sched_group_barrier(0x008, 1, 0); // 1 WMMA
            __builtin_amdgcn_sched_group_barrier(0x100, 2, 0); // next fragment DS reads
        }
        __builtin_amdgcn_sched_group_barrier(0x008, 1, 0);     // final WMMA

        // epilogue: s += exp2(C*(cos-1))  (fixed max = 20, logits bounded)
#pragma unroll
        for (int g = 0; g < 8; ++g) {
            const float x = fmaf(acc[g], C20L2E, -C20L2E);
            s[g] += __builtin_amdgcn_exp2f(x);
        }

        asm volatile("s_wait_asynccnt 0x0" ::: "memory");
        __syncthreads();   // publish prefetched buffer; release consumed buffer
    }

    // reduce each row's partial sum over the 16 lanes of its half-wave
#pragma unroll
    for (int g = 0; g < 8; ++g) {
        float v = s[g];
        v += __shfl_xor(v, 1);
        v += __shfl_xor(v, 2);
        v += __shfl_xor(v, 4);
        v += __shfl_xor(v, 8);
        if (r == g) ssum[mbase + g + 8 * h] = v;   // lanes g and g+16 write rows g, g+8
    }
}

// ---------------------------------------------------------------------------
// Kernel 3: loss = mean_i( (20 + ln(ssum_i)) - diag_i )
// ---------------------------------------------------------------------------
__global__ void k_finalize(const float* __restrict__ ssum,
                           const float* __restrict__ diag,
                           float* __restrict__ out, int N) {
    float acc = 0.f;
    for (int i = threadIdx.x; i < N; i += blockDim.x)
        acc += (20.0f + logf(ssum[i])) - diag[i];
#pragma unroll
    for (int m = 1; m < 32; m <<= 1) acc += __shfl_xor(acc, m);
    __shared__ float sm[8];
    const int wv = threadIdx.x >> 5;
    const int ln = threadIdx.x & 31;
    if (ln == 0) sm[wv] = acc;
    __syncthreads();
    if (threadIdx.x == 0) {
        float t = 0.f;
#pragma unroll
        for (int w = 0; w < 8; ++w) t += sm[w];
        out[0] = t / (float)N;
    }
}

// ---------------------------------------------------------------------------
extern "C" void kernel_launch(void* const* d_in, const int* in_sizes, int n_in,
                              void* d_out, int out_size, void* d_ws, size_t ws_size,
                              hipStream_t stream) {
    const float* in = (const float*)d_in[0];
    const int N = in_sizes[0] / (2 * D);   // 8192

    char* ws = (char*)d_ws;
    _Float16* n1h = (_Float16*)ws;
    _Float16* n2h = n1h + (size_t)N * D;
    float*    diag = (float*)(n2h + (size_t)N * D);
    float*    ssum = diag + N;
    float*    out  = (float*)d_out;

    (void)hipFuncSetAttribute((const void*)k_gemm_lse,
                              hipFuncAttributeMaxDynamicSharedMemorySize,
                              LDS_BYTES);

    k_normalize<<<N, 256, 0, stream>>>(in, n1h, n2h, diag);

    const int mtiles = N / 16;             // 512
    k_gemm_lse<<<mtiles / WPB, WPB * 32, LDS_BYTES, stream>>>(n1h, n2h, ssum, mtiles);

    k_finalize<<<1, 256, 0, stream>>>(ssum, diag, out, N);
}
